// Mute_78125455115021
// MI455X (gfx1250) — matrix operand order repaired and checked by reading, not compiled, not run on hardware
//
#include <hip/hip_runtime.h>

// MI455X / gfx1250, wave32. Fused attention-mask pipeline:
//   scores (the 268MB intermediate) never leaves registers; bf16 WMMA with
//   f32 accumulation for all GEMMs; softmax-over-batch fused into the score
//   GEMM as a running column-max of (s - logsumexp_b(s)).
//   Shared Qt tiles staged to LDS with double-buffered async copies
//   (ASYNCcnt), A-operands served by ds_load_b128 instead of 8x-redundant L2.

#define BN  4
#define CN  256
#define HWN 4096
#define NCHUNK 8
#define SCALE_QK 0.0625f   // 1/sqrt(256)

typedef __attribute__((ext_vector_type(16))) __bf16 v16bf;
typedef __attribute__((ext_vector_type(8)))  float  v8f;

typedef int v4i __attribute__((__vector_size__(16)));
typedef __attribute__((address_space(1))) v4i gv4i;   // global-AS 16B vector
typedef __attribute__((address_space(3))) v4i lv4i;   // LDS-AS 16B vector

#if __has_builtin(__builtin_amdgcn_global_load_async_to_lds_b128) && \
    __has_builtin(__builtin_amdgcn_s_wait_asynccnt)
#define ASYNC_OK 1
#define WAIT_ASYNC(n) __builtin_amdgcn_s_wait_asynccnt(n)
#else
#define ASYNC_OK 0
#define WAIT_ASYNC(n)
#endif

union ABPack { v16bf v; uint4 u[2]; };

static __device__ __forceinline__ v8f wmma_bf16(v16bf a, v16bf b, v8f c) {
  // D = A(16x32 bf16) x B(32x16 bf16) + C(16x16 f32)
  return __builtin_amdgcn_wmma_f32_16x16x32_bf16(
      /*neg_a=*/false, a, /*neg_b=*/false, b,
      /*c_mod=*/(short)0, c, /*reuse_a=*/false, /*reuse_b=*/false);
}

// Stage one Qt n-tile (all 4 batches: 4*16 rows x 256 bf16 = 32KB) into LDS.
// 2048 16-byte chunks spread over 256 threads -> 8 async b128 copies each.
static __device__ __forceinline__ void stage_q_tile(const __bf16* __restrict__ Qt,
                                                    __bf16* qs_buf, int n0, int tid) {
  #pragma unroll
  for (int j = 0; j < 8; ++j) {
    int ch = j * 256 + tid;            // 0..2047
    int b  = ch >> 9;                  // 512 chunks per batch
    int r  = (ch >> 5) & 15;           // row within tile
    int c8 = (ch & 31) * 8;            // bf16 column (8 bf16 = 16B chunk)
    const __bf16* g = Qt + ((size_t)b * HWN + n0 + r) * CN + c8;
    __bf16*       l = qs_buf + ((b * 16 + r) * CN + c8);
#if ASYNC_OK
    __builtin_amdgcn_global_load_async_to_lds_b128(
        (gv4i*)(const void*)g, (lv4i*)(void*)l, 0, 0);
#else
    *(uint4*)l = *(const uint4*)g;     // fallback: sync copy via VGPRs
#endif
  }
}

// ---------------------------------------------------------------- prep: X -> Xt (bf16, [b][n][c])
__global__ void prep_x(const float* __restrict__ fuse, __bf16* __restrict__ Xt) {
  int idx = blockIdx.x * 256 + threadIdx.x;       // BN*HWN*CN = 4194304
  int c = idx & 255;
  int n = (idx >> 8) & 4095;
  int b = idx >> 20;
  Xt[idx] = (__bf16)fuse[((size_t)b * CN + c) * HWN + n];
}

// ---------------------------------------------------------------- prep: W{q,k,v} -> bf16
__global__ void prep_w(const float* __restrict__ Wq, const float* __restrict__ Wk,
                       const float* __restrict__ Wv, __bf16* __restrict__ Wbf) {
  int idx = blockIdx.x * 256 + threadIdx.x;       // 3*256*256 = 196608
  int m = idx >> 16;
  int r = idx & 65535;
  const float* src = (m == 0) ? Wq : ((m == 1) ? Wk : Wv);
  Wbf[idx] = (__bf16)src[r];
}

// ---------------------------------------------------------------- phase 1: Q,K,V GEMMs
// Each wave: one (batch, matrix, 16-row o-tile, 64-wide n-strip).
// Q,K written transposed bf16 [b][n][c]; V written plain f32 [b][c][n].
__global__ void qkv_gemm(const __bf16* __restrict__ Xt, const __bf16* __restrict__ Wbf,
                         const float* __restrict__ bq, const float* __restrict__ bk,
                         const float* __restrict__ bv,
                         __bf16* __restrict__ Qt, __bf16* __restrict__ Kt,
                         float* __restrict__ V) {
  const int lane  = threadIdx.x & 31;
  const int laneM = lane & 15;
  const int hi    = lane >> 4;
  int wid = blockIdx.x * (blockDim.x >> 5) + (threadIdx.x >> 5);  // 0..12287
  const int b   = wid / 3072;
  int rem = wid - b * 3072;
  const int mat = rem >> 10;             // 0=Q 1=K 2=V
  rem &= 1023;
  const int o0  = (rem >> 6) << 4;       // output-channel tile base
  const int nb  = (rem & 63) << 6;       // 64-wide n strip base

  const __bf16* Wm   = Wbf + (size_t)mat * CN * CN;
  const float*  bias = (mat == 0) ? bq : ((mat == 1) ? bk : bv);

  float bias8[8];
  #pragma unroll
  for (int r = 0; r < 8; ++r) bias8[r] = bias[o0 + r + 8 * hi];

  v8f acc[4] = {};
  const __bf16* wrow  = Wm + (size_t)(o0 + laneM) * CN;
  const __bf16* xbase = Xt + (size_t)b * HWN * CN;

  for (int k0 = 0; k0 < CN; k0 += 32) {
    ABPack a;                                        // A: W tile 16x32 (row-major rows contiguous)
    a.u[0] = *(const uint4*)(wrow + k0 + hi * 16);
    a.u[1] = *(const uint4*)(wrow + k0 + hi * 16 + 8);
    if (k0 + 32 < CN)                                // gfx1250 global_prefetch_b8
      __builtin_prefetch(xbase + (size_t)(nb + laneM) * CN + k0 + 32, 0, 3);
    #pragma unroll
    for (int t = 0; t < 4; ++t) {                    // B: Xt tile 32x16 (per-lane K contiguous)
      const __bf16* xr = xbase + (size_t)(nb + t * 16 + laneM) * CN + k0 + hi * 16;
      ABPack bm;
      bm.u[0] = ((const uint4*)xr)[0];
      bm.u[1] = ((const uint4*)xr)[1];
      acc[t] = wmma_bf16(a.v, bm.v, acc[t]);
    }
  }

  #pragma unroll
  for (int t = 0; t < 4; ++t) {
    const int n0 = nb + t * 16;
    if (mat < 2) {
      __bf16* dst = (mat == 0 ? Qt : Kt) + ((size_t)b * HWN + n0 + laneM) * CN + o0 + 8 * hi;
      union { uint4 u; __bf16 h[8]; } st;
      #pragma unroll
      for (int r = 0; r < 8; ++r) st.h[r] = (__bf16)(acc[t][r] + bias8[r]);
      *(uint4*)dst = st.u;                           // packed 16B transposed store
    } else {
      #pragma unroll
      for (int r = 0; r < 8; ++r)
        V[((size_t)b * CN + o0 + r + 8 * hi) * HWN + n0 + laneM] = acc[t][r] + bias8[r];
    }
  }
}

// ---------------------------------------------------------------- Q_mask[(c%4)][n] = max over b, c/4
__global__ void qmask_k(const __bf16* __restrict__ Qt, float* __restrict__ Qmask) {
  int tid = blockIdx.x * 256 + threadIdx.x;        // 16384
  int n  = tid & 4095;
  int cl = tid >> 12;
  float v = -3.0e38f;
  #pragma unroll
  for (int b = 0; b < BN; ++b) {
    const __bf16* row = Qt + ((size_t)b * HWN + n) * CN + cl;
    for (int ch = 0; ch < 64; ++ch) v = fmaxf(v, (float)row[ch * 4]);
  }
  Qmask[(size_t)cl * HWN + n] = v;
}

// ---------------------------------------------------------------- phase 2: fused score GEMM +
// batch-softmax + running max over n.  pcol[chunk][b][m] = max_n (s[b,n,m] - lse_b)
// Qt tiles (shared by all 8 waves) are double-buffered in LDS via async copies.
__global__ void __launch_bounds__(256) attn_colmax(const __bf16* __restrict__ Qt,
                                                   const __bf16* __restrict__ Kt,
                                                   float* __restrict__ pcol) {
  __shared__ __bf16 qs[2][BN * 16 * CN];            // 2 x 32KB double buffer

  const int tid   = threadIdx.x;
  const int lane  = tid & 31;
  const int laneM = lane & 15;
  const int hi    = lane >> 4;
  const int wv    = tid >> 5;                       // wave in block, 0..7
  const int m0    = (blockIdx.x * 8 + wv) * 16;     // key-tile base (per wave)
  const int chunk = blockIdx.y;                     // 0..NCHUNK-1

  float rmax0 = -3.0e38f, rmax1 = -3.0e38f, rmax2 = -3.0e38f, rmax3 = -3.0e38f;

  const int nt0 = chunk * 32, nt1 = nt0 + 32;
  stage_q_tile(Qt, qs[0], nt0 * 16, tid);           // preload first tile

  for (int nt = nt0; nt < nt1; ++nt) {
    const int buf = (nt - nt0) & 1;
    if (nt + 1 < nt1) {
      stage_q_tile(Qt, qs[buf ^ 1], (nt + 1) * 16, tid);  // prefetch next tile
      WAIT_ASYNC(8);                                // drain current tile, keep next in flight
    } else {
      WAIT_ASYNC(0);
    }
    __syncthreads();                                // all waves' LDS writes visible

    v8f acc[4] = {};
    const __bf16* q0 = &qs[buf][(0 * 16 + laneM) * CN];
    const __bf16* q1 = &qs[buf][(1 * 16 + laneM) * CN];
    const __bf16* q2 = &qs[buf][(2 * 16 + laneM) * CN];
    const __bf16* q3 = &qs[buf][(3 * 16 + laneM) * CN];
    const __bf16* qrow[4] = {q0, q1, q2, q3};

    for (int k0 = 0; k0 < CN; k0 += 32) {
      #pragma unroll
      for (int b = 0; b < BN; ++b) {
        ABPack a;                                   // A = Q^T tile from LDS (ds_load_b128)
        a.u[0] = *(const uint4*)(qrow[b] + k0 + hi * 16);
        a.u[1] = *(const uint4*)(qrow[b] + k0 + hi * 16 + 8);
        const __bf16* kr = Kt + ((size_t)b * HWN + m0 + laneM) * CN + k0 + hi * 16;
        ABPack bm;                                  // B = K tile from L2 (per-wave private)
        bm.u[0] = ((const uint4*)kr)[0];
        bm.u[1] = ((const uint4*)kr)[1];
        acc[b] = wmma_bf16(a.v, bm.v, acc[b]);
      }
    }
    #pragma unroll
    for (int i = 0; i < 8; ++i) {
      float s0 = acc[0][i] * SCALE_QK, s1 = acc[1][i] * SCALE_QK;
      float s2 = acc[2][i] * SCALE_QK, s3 = acc[3][i] * SCALE_QK;
      float mx  = fmaxf(fmaxf(s0, s1), fmaxf(s2, s3));
      float lse = mx + __logf(__expf(s0 - mx) + __expf(s1 - mx) +
                              __expf(s2 - mx) + __expf(s3 - mx));
      rmax0 = fmaxf(rmax0, s0 - lse);
      rmax1 = fmaxf(rmax1, s1 - lse);
      rmax2 = fmaxf(rmax2, s2 - lse);
      rmax3 = fmaxf(rmax3, s3 - lse);
    }
    __syncthreads();                                // protect buffer reuse
  }

  // lanes L and L+16 hold the same column m -> combine halves (wave32)
  rmax0 = fmaxf(rmax0, __shfl_xor(rmax0, 16, 32));
  rmax1 = fmaxf(rmax1, __shfl_xor(rmax1, 16, 32));
  rmax2 = fmaxf(rmax2, __shfl_xor(rmax2, 16, 32));
  rmax3 = fmaxf(rmax3, __shfl_xor(rmax3, 16, 32));
  if (hi == 0) {
    float r[4] = {rmax0, rmax1, rmax2, rmax3};
    #pragma unroll
    for (int b = 0; b < BN; ++b)
      pcol[((size_t)chunk * BN + b) * HWN + m0 + laneM] = r[b];
  }
}

// ---------------------------------------------------------------- phase 3: m2[b*64+y] = exp(max_x colmax)
__global__ void reduce_m2(const float* __restrict__ pcol, float* __restrict__ m2) {
  int tid = threadIdx.x;                 // 256
  int b = tid >> 6, y = tid & 63;
  float v = -3.0e38f;
  for (int x = 0; x < 64; ++x) {
    int mm = y * 64 + x;
    #pragma unroll
    for (int ch = 0; ch < NCHUNK; ++ch)
      v = fmaxf(v, pcol[((size_t)ch * BN + b) * HWN + mm]);
  }
  m2[tid] = __expf(v);
}

// ---------------------------------------------------------------- phase 4: out = V*(1 + m2*Qmask)
__global__ void finalize_k(const float* __restrict__ V, const float* __restrict__ m2,
                           const float* __restrict__ Qmask, float* __restrict__ out) {
  int idx = blockIdx.x * 256 + threadIdx.x;    // 4194304
  int n = idx & 4095;
  int c = (idx >> 12) & 255;
  int b = idx >> 20;
  float mask = m2[b * 64 + (c >> 2)] * Qmask[(size_t)(c & 3) * HWN + n];
  float v = V[idx];
  out[idx] = v + v * mask;
}

// ---------------------------------------------------------------- launch
extern "C" void kernel_launch(void* const* d_in, const int* in_sizes, int n_in,
                              void* d_out, int out_size, void* d_ws, size_t ws_size,
                              hipStream_t stream) {
  const float* fuse = (const float*)d_in[0];
  const float* Wq   = (const float*)d_in[1];
  const float* bq   = (const float*)d_in[2];
  const float* Wk   = (const float*)d_in[3];
  const float* bk   = (const float*)d_in[4];
  const float* Wv   = (const float*)d_in[5];
  const float* bv   = (const float*)d_in[6];
  float* out = (float*)d_out;

  char* ws = (char*)d_ws;
  size_t offXt = 0;                                        // bf16 [4][4096][256]  8 MB
  size_t offWb = offXt + (size_t)BN * HWN * CN * 2;        // bf16 [3][256][256]   384 KB
  size_t offQt = offWb + (size_t)3 * CN * CN * 2;          // bf16 [4][4096][256]  8 MB
  size_t offKt = offQt + (size_t)BN * HWN * CN * 2;        // bf16 [4][4096][256]  8 MB
  size_t offV  = offKt + (size_t)BN * HWN * CN * 2;        // f32  [4][256][4096] 16 MB
  size_t offQm = offV  + (size_t)BN * CN * HWN * 4;        // f32  [4][4096]       64 KB
  size_t offPc = offQm + (size_t)4 * HWN * 4;              // f32  [8][4][4096]   512 KB
  size_t offM2 = offPc + (size_t)NCHUNK * BN * HWN * 4;    // f32  [256]

  __bf16* Xt  = (__bf16*)(ws + offXt);
  __bf16* Wbf = (__bf16*)(ws + offWb);
  __bf16* Qt  = (__bf16*)(ws + offQt);
  __bf16* Kt  = (__bf16*)(ws + offKt);
  float*  V   = (float*) (ws + offV);
  float*  Qm  = (float*) (ws + offQm);
  float*  Pc  = (float*) (ws + offPc);
  float*  M2  = (float*) (ws + offM2);

  prep_x<<<16384, 256, 0, stream>>>(fuse, Xt);
  prep_w<<<768,   256, 0, stream>>>(Wq, Wk, Wv, Wbf);
  qkv_gemm<<<1536, 256, 0, stream>>>(Xt, Wbf, bq, bk, bv, Qt, Kt, V);   // 12288 waves
  qmask_k<<<64,   256, 0, stream>>>(Qt, Qm);
  attn_colmax<<<dim3(32, NCHUNK), 256, 0, stream>>>(Qt, Kt, Pc);        // 2048 waves
  reduce_m2<<<1,  256, 0, stream>>>(Pc, M2);
  finalize_k<<<16384, 256, 0, stream>>>(V, M2, Qm, out);
}
